// HiSACKT_NoSkillGroup_11046655885797
// MI455X (gfx1250) — matrix operand order repaired and checked by reading, not compiled, not run on hardware
//
#include <hip/hip_runtime.h>
#include <hip/hip_bf16.h>
#include <stdint.h>

// ---------------------------------------------------------------------------
// CDNA5 (gfx1250) transformer pipeline. All matmuls via v_wmma_f32_16x16x32_bf16.
// Double-buffered LDS staging via GLOBAL_LOAD_ASYNC_TO_LDS_B128 when available.
// wave32 everywhere. f32 master activations + bf16 GEMM operands.
// ---------------------------------------------------------------------------

typedef __attribute__((ext_vector_type(16))) __bf16 v16bf;
typedef __attribute__((ext_vector_type(8)))  float  v8f;

#if defined(__AMDGCN__) && __has_builtin(__builtin_amdgcn_global_load_async_to_lds_b128)
#define HAS_ASYNC_LDS 1
typedef int v4i_async __attribute__((__vector_size__(4 * sizeof(int))));
#else
#define HAS_ASYNC_LDS 0
#endif

__device__ inline void wait_async_lds() {
#if HAS_ASYNC_LDS
#if __has_builtin(__builtin_amdgcn_s_wait_asynccnt)
  __builtin_amdgcn_s_wait_asynccnt(0);
#else
  asm volatile("s_wait_asynccnt 0x0" ::: "memory");
#endif
#endif
}

union BV { uint32_t u[8]; v16bf v; };

// A operand: 16x32 bf16 tile, row-major (stride lda). ISA layout:
//   lanes 0-15 (kh=0): VGPR0-3 K=0..7, VGPR4-7 K=16..23  (row = lane)
//   lanes 16-31 (kh=1): VGPR0-3 K=8..15, VGPR4-7 K=24..31
__device__ inline v16bf ld_a16x32(const __bf16* base, int lda, int lane) {
  const int row = lane & 15, kh = (lane >> 4) & 1;
  const uint32_t* p = (const uint32_t*)(base + row * lda + kh * 8);
  BV t;
  t.u[0] = p[0]; t.u[1] = p[1]; t.u[2] = p[2]; t.u[3] = p[3];
  t.u[4] = p[8]; t.u[5] = p[9]; t.u[6] = p[10]; t.u[7] = p[11];
  return t.v;
}

// B operand: 32x16 bf16 (KxN). W held row-major [N][K] (so B = W^T):
//   lanes 0-15: column = lane, VGPR0-3 K=0..7, VGPR4-7 K=16..23
//   lanes 16-31: column = lane-16, VGPR0-3 K=8..15, VGPR4-7 K=24..31
__device__ inline v16bf ld_b32x16(const __bf16* base, int ldw, int lane) {
  const int col = lane & 15, kh = (lane >> 4) & 1;
  const uint32_t* p = (const uint32_t*)(base + col * ldw + kh * 8);
  BV t;
#pragma unroll
  for (int i = 0; i < 8; ++i) t.u[i] = p[(i & 3) + ((i >> 2) << 3)];
  return t.v;
}

// Stage one 64x32 bf16 tile (row-major, stride ld) into LDS. 128 threads copy
// 256 x 16B chunks (2 per thread). Async to LDS when available.
__device__ inline void stage_tile(const __bf16* __restrict__ g, int ld, int rbase,
                                  int k0, __bf16* lds)
{
#pragma unroll
  for (int t = 0; t < 2; ++t) {
    int ci = (int)threadIdx.x + t * 128;     // 0..255
    int row = ci >> 2, q = ci & 3;
    const __bf16* src = g + (size_t)(rbase + row) * ld + k0 + q * 8;
    __bf16* dst = lds + row * 32 + q * 8;
#if HAS_ASYNC_LDS
    __builtin_amdgcn_global_load_async_to_lds_b128((v4i_async*)src, (v4i_async*)dst, 0, 0);
#else
    *(uint4*)dst = *(const uint4*)src;
#endif
  }
}

// ---------------------------------------------------------------------------
// Batched GEMM: C[M x Nc] = scale * (A[M x K] @ W[Nc x K]^T) + bias (+ReLU)
// 128 threads = 4 waves in a 2x2 grid; block tile 64x64; wave tile 32x32;
// K step 32; double-buffered LDS.
// ---------------------------------------------------------------------------
__global__ __launch_bounds__(128)
void k_gemm(const __bf16* __restrict__ A, int lda, unsigned long long sA,
            const __bf16* __restrict__ W, int ldw, unsigned long long sW,
            float* __restrict__ C, int ldc, unsigned long long sC,
            const float* __restrict__ bias, float scale, int relu, int K)
{
  __shared__ __align__(16) __bf16 As[2][64 * 32];
  __shared__ __align__(16) __bf16 Ws[2][64 * 32];
  A += (unsigned long long)blockIdx.z * sA;
  W += (unsigned long long)blockIdx.z * sW;
  C += (unsigned long long)blockIdx.z * sC;
  const int bm = blockIdx.x * 64, bn = blockIdx.y * 64;
  const int wave = threadIdx.x >> 5, lane = threadIdx.x & 31;
  const int wm = wave >> 1, wn = wave & 1;
  v8f acc[2][2] = {};

  int buf = 0;
  stage_tile(A, lda, bm, 0, &As[0][0]);
  stage_tile(W, ldw, bn, 0, &Ws[0][0]);

  for (int k0 = 0; k0 < K; k0 += 32) {
    wait_async_lds();      // this wave's copies into As[buf]/Ws[buf] complete
    __syncthreads();       // all waves' copies complete; prior reads of buf^1 done
    if (k0 + 32 < K) {
      stage_tile(A, lda, bm, k0 + 32, &As[buf ^ 1][0]);
      stage_tile(W, ldw, bn, k0 + 32, &Ws[buf ^ 1][0]);
    }
    v16bf a0 = ld_a16x32(&As[buf][(wm * 32) * 32], 32, lane);
    v16bf a1 = ld_a16x32(&As[buf][(wm * 32 + 16) * 32], 32, lane);
    v16bf b0 = ld_b32x16(&Ws[buf][(wn * 32) * 32], 32, lane);
    v16bf b1 = ld_b32x16(&Ws[buf][(wn * 32 + 16) * 32], 32, lane);
    acc[0][0] = __builtin_amdgcn_wmma_f32_16x16x32_bf16(false, a0, false, b0, (short)0, acc[0][0], false, false);
    acc[0][1] = __builtin_amdgcn_wmma_f32_16x16x32_bf16(false, a0, false, b1, (short)0, acc[0][1], false, false);
    acc[1][0] = __builtin_amdgcn_wmma_f32_16x16x32_bf16(false, a1, false, b0, (short)0, acc[1][0], false, false);
    acc[1][1] = __builtin_amdgcn_wmma_f32_16x16x32_bf16(false, a1, false, b1, (short)0, acc[1][1], false, false);
    buf ^= 1;
  }

  // C/D layout: lane&15 = column, VGPR r -> row r (lanes<16) / r+8 (lanes>=16)
#pragma unroll
  for (int mi = 0; mi < 2; ++mi) {
#pragma unroll
    for (int ni = 0; ni < 2; ++ni) {
      const int r0 = bm + wm * 32 + mi * 16 + ((lane >> 4) << 3);
      const int col = bn + wn * 32 + ni * 16 + (lane & 15);
      float bv = bias ? bias[col] : 0.f;
#pragma unroll
      for (int r = 0; r < 8; ++r) {
        float v = acc[mi][ni][r] * scale + bv;
        if (relu) v = fmaxf(v, 0.f);
        C[(size_t)(r0 + r) * ldc + col] = v;
      }
    }
  }
}

// ---------------------------------------------------------------------------
// Elementwise / reduction helper kernels
// ---------------------------------------------------------------------------
__global__ void k_f32_to_bf16(const float* __restrict__ x, __bf16* __restrict__ y, size_t n) {
  size_t i = (size_t)blockIdx.x * blockDim.x + threadIdx.x;
  if (i < n) y[i] = (__bf16)x[i];
}

__global__ void k_zero_u32(uint32_t* p, size_t n) {
  size_t i = (size_t)blockIdx.x * blockDim.x + threadIdx.x;
  if (i < n) p[i] = 0u;
}

// M = M_emb[q + NQ*r] + P ; E = E_emb[qry]   (f32 + bf16 copies)
__global__ void k_embed(const int* __restrict__ q, const int* __restrict__ r,
                        const int* __restrict__ qry,
                        const float* __restrict__ M_emb, const float* __restrict__ E_emb,
                        const float* __restrict__ P,
                        float* __restrict__ Mf, __bf16* __restrict__ Mb,
                        float* __restrict__ Ef, __bf16* __restrict__ Eb)
{
  size_t idx = (size_t)blockIdx.x * blockDim.x + threadIdx.x;  // over 8192*512
  if (idx >= (size_t)8192 * 512) return;
  int row = (int)(idx >> 9), c = (int)(idx & 511);
  int n = row & 511;
  int x = q[row] + 10000 * r[row];
  float m = M_emb[(size_t)x * 512 + c] + P[(size_t)n * 512 + c];
  float e = E_emb[(size_t)qry[row] * 512 + c];
  Mf[idx] = m; Mb[idx] = (__bf16)m;
  Ef[idx] = e; Eb[idx] = (__bf16)e;
}

// QKV[rows,1536] f32 -> Q,K bf16 [B,H,Nq,64], Vt bf16 [B,H,64,Nq]
__global__ void k_split_heads(const float* __restrict__ QKV,
                              __bf16* __restrict__ Qb, __bf16* __restrict__ Kb,
                              __bf16* __restrict__ Vt, int Nq, int rows)
{
  size_t idx = (size_t)blockIdx.x * blockDim.x + threadIdx.x;  // rows*512
  if (idx >= (size_t)rows * 512) return;
  int row = (int)(idx >> 9), c = (int)(idx & 511);
  int b = row / Nq, n = row % Nq;
  int h = c >> 6, e = c & 63;
  size_t base = (size_t)row * 1536;
  size_t qi = (((size_t)(b * 8 + h) * Nq) + n) * 64 + e;
  Qb[qi] = (__bf16)QKV[base + c];
  Kb[qi] = (__bf16)QKV[base + 512 + c];
  Vt[(((size_t)(b * 8 + h)) * 64 + e) * (size_t)Nq + n] = (__bf16)QKV[base + 1024 + c];
}

// PV [B,H,Nq,64] f32 -> Acat bf16 [rows,512]
__global__ void k_merge_heads(const float* __restrict__ PV, __bf16* __restrict__ Acat,
                              int Nq, int rows)
{
  size_t idx = (size_t)blockIdx.x * blockDim.x + threadIdx.x;
  if (idx >= (size_t)rows * 512) return;
  int row = (int)(idx >> 9), c = (int)(idx & 511);
  int b = row / Nq, n = row % Nq;
  int h = c >> 6, e = c & 63;
  Acat[idx] = (__bf16)PV[(((size_t)(b * 8 + h) * Nq) + n) * 64 + e];
}

// Row softmax over Nk f32 scores; writes bf16 probabilities in place at row start.
__global__ void k_softmax(float* __restrict__ S, int Nk, int Nq, int causal)
{
  __shared__ float red[8];
  __shared__ float bcast;
  const int row = blockIdx.x, tid = threadIdx.x, bs = blockDim.x;
  const int q = row % Nq;
  float* Srow = S + (size_t)row * Nk;
  const int nv = Nk / bs;
  float vals[8];
  float m = -3.4e38f;
  for (int j = 0; j < nv; ++j) {
    int k = tid + j * bs;
    float v = Srow[k];
    bool ok = (!causal) || (k <= q);
    vals[j] = ok ? v : -3.4e38f;
    m = fmaxf(m, vals[j]);
  }
  for (int o = 16; o; o >>= 1) m = fmaxf(m, __shfl_xor(m, o, 32));
  const int nw = bs >> 5;
  if ((tid & 31) == 0) red[tid >> 5] = m;
  __syncthreads();
  if (tid == 0) { float t = red[0]; for (int i = 1; i < nw; ++i) t = fmaxf(t, red[i]); bcast = t; }
  __syncthreads();
  m = bcast;
  float s = 0.f;
  for (int j = 0; j < nv; ++j) {
    float e = (vals[j] > -1.0e37f) ? __expf(vals[j] - m) : 0.f;
    vals[j] = e; s += e;
  }
  for (int o = 16; o; o >>= 1) s += __shfl_xor(s, o, 32);
  if ((tid & 31) == 0) red[tid >> 5] = s;
  __syncthreads();
  if (tid == 0) { float t = 0.f; for (int i = 0; i < nw; ++i) t += red[i]; bcast = t; }
  __syncthreads();
  float inv = 1.f / bcast;
  __bf16* Prow = (__bf16*)Srow;  // in-place, bf16 row stride = 2*Nk
  for (int j = 0; j < nv; ++j) Prow[tid + j * bs] = (__bf16)(vals[j] * inv);
}

// attn_w[b,q,k] = (1/8) * sum_h P_bf16[(b*8+h)*512+q, k]   (P stride 1024 bf16)
__global__ void k_attn_mean(const __bf16* __restrict__ Pall, float* __restrict__ out)
{
  size_t idx = (size_t)blockIdx.x * blockDim.x + threadIdx.x;  // 16*512*512
  if (idx >= (size_t)16 * 512 * 512) return;
  int k = (int)(idx & 511);
  int q = (int)((idx >> 9) & 511);
  int b = (int)(idx >> 18);
  float s = 0.f;
#pragma unroll
  for (int h = 0; h < 8; ++h)
    s += (float)Pall[((size_t)((b * 8 + h) * 512 + q)) * 1024 + k];
  out[idx] = s * 0.125f;
}

// y = LN(x1 + x2 [+ x3]) * g + b ; optional bf16 copy. One block per 512-row.
__global__ __launch_bounds__(256)
void k_add_ln(const float* __restrict__ x1, const float* __restrict__ x2,
              const float* __restrict__ x3, const float* __restrict__ g,
              const float* __restrict__ bt, float* __restrict__ y,
              __bf16* __restrict__ ybf)
{
  __shared__ float red[8];
  __shared__ float mean_s, rstd_s;
  const int row = blockIdx.x, tid = threadIdx.x;
  const size_t base = (size_t)row * 512;
  float v0 = x1[base + tid] + x2[base + tid];
  float v1 = x1[base + tid + 256] + x2[base + tid + 256];
  if (x3) { v0 += x3[base + tid]; v1 += x3[base + tid + 256]; }
  float s = v0 + v1;
  for (int o = 16; o; o >>= 1) s += __shfl_xor(s, o, 32);
  if ((tid & 31) == 0) red[tid >> 5] = s;
  __syncthreads();
  if (tid == 0) { float t = 0.f; for (int i = 0; i < 8; ++i) t += red[i]; mean_s = t * (1.f / 512.f); }
  __syncthreads();
  float m = mean_s;
  float d0 = v0 - m, d1 = v1 - m;
  float sq = d0 * d0 + d1 * d1;
  for (int o = 16; o; o >>= 1) sq += __shfl_xor(sq, o, 32);
  if ((tid & 31) == 0) red[tid >> 5] = sq;
  __syncthreads();
  if (tid == 0) { float t = 0.f; for (int i = 0; i < 8; ++i) t += red[i]; rstd_s = rsqrtf(t * (1.f / 512.f) + 1e-5f); }
  __syncthreads();
  float rs = rstd_s;
  float o0 = d0 * rs * g[tid] + bt[tid];
  float o1 = d1 * rs * g[tid + 256] + bt[tid + 256];
  y[base + tid] = o0; y[base + tid + 256] = o1;
  if (ybf) { ybf[base + tid] = (__bf16)o0; ybf[base + tid + 256] = (__bf16)o1; }
}

__global__ void k_add4(const float* a, const float* b, const float* c, const float* d,
                       float* y, __bf16* ybf, size_t n)
{
  size_t i = (size_t)blockIdx.x * blockDim.x + threadIdx.x;
  if (i < n) { float v = a[i] + b[i] + c[i] + d[i]; y[i] = v; ybf[i] = (__bf16)v; }
}

// gather last-64 rows of S_base_bf16 into packed [16*64, 512]
__global__ void k_pack_loc(const __bf16* __restrict__ Sb, __bf16* __restrict__ locA)
{
  size_t idx = (size_t)blockIdx.x * blockDim.x + threadIdx.x;  // 16*64*512
  if (idx >= (size_t)16 * 64 * 512) return;
  int c = (int)(idx & 511);
  int i = (int)((idx >> 9) & 63);
  int b = (int)(idx >> 15);
  locA[idx] = Sb[((size_t)(b * 512 + 448 + i)) * 512 + c];
}

__global__ void k_scatter_loc(const float* __restrict__ locOut,
                              float* __restrict__ Sl, __bf16* __restrict__ Slb)
{
  size_t idx = (size_t)blockIdx.x * blockDim.x + threadIdx.x;
  if (idx >= (size_t)16 * 64 * 512) return;
  int c = (int)(idx & 511);
  int i = (int)((idx >> 9) & 63);
  int b = (int)(idx >> 15);
  size_t dst = ((size_t)(b * 512 + 448 + i)) * 512 + c;
  float v = locOut[idx];
  Sl[dst] = v; Slb[dst] = (__bf16)v;
}

// p[row] = sigmoid(dot(F[row], w) + b). One wave per row.
__global__ __launch_bounds__(256)
void k_pred(const float* __restrict__ F, const float* __restrict__ w,
            const float* __restrict__ b, float* __restrict__ out)
{
  int row = blockIdx.x * 8 + (threadIdx.x >> 5);
  int lane = threadIdx.x & 31;
  const float* Fr = F + (size_t)row * 512;
  float s = 0.f;
  for (int c = lane; c < 512; c += 32) s += Fr[c] * w[c];
  for (int o = 16; o; o >>= 1) s += __shfl_xor(s, o, 32);
  if (lane == 0) out[row] = 1.f / (1.f + __expf(-(s + b[0])));
}

// ---------------------------------------------------------------------------
// Host orchestration
// ---------------------------------------------------------------------------
extern "C" void kernel_launch(void* const* d_in, const int* in_sizes, int n_in,
                              void* d_out, int out_size, void* d_ws, size_t ws_size,
                              hipStream_t stream)
{
  (void)in_sizes; (void)n_in; (void)out_size; (void)ws_size;
  const int*   qI    = (const int*)d_in[0];
  const int*   rI    = (const int*)d_in[1];
  const int*   qryI  = (const int*)d_in[2];
  const float* M_emb = (const float*)d_in[3];
  const float* E_emb = (const float*)d_in[4];
  const float* Pmat  = (const float*)d_in[5];
  const float* in_w[4]  = {(const float*)d_in[6],  (const float*)d_in[10], (const float*)d_in[14], (const float*)d_in[18]};
  const float* in_b[4]  = {(const float*)d_in[7],  (const float*)d_in[11], (const float*)d_in[15], (const float*)d_in[19]};
  const float* out_w[4] = {(const float*)d_in[8],  (const float*)d_in[12], (const float*)d_in[16], (const float*)d_in[20]};
  const float* out_b[4] = {(const float*)d_in[9],  (const float*)d_in[13], (const float*)d_in[17], (const float*)d_in[21]};
  const float* ln1_g = (const float*)d_in[22];
  const float* ln1_b = (const float*)d_in[23];
  const float* w1    = (const float*)d_in[24];
  const float* b1    = (const float*)d_in[25];
  const float* w2    = (const float*)d_in[26];
  const float* b2    = (const float*)d_in[27];
  const float* ln2_g = (const float*)d_in[28];
  const float* ln2_b = (const float*)d_in[29];
  const float* predw = (const float*)d_in[30];
  const float* predb = (const float*)d_in[31];
  float* outF = (float*)d_out;

  const size_t R = 8192, RD = (size_t)8192 * 512;
  char* ws = (char*)d_ws;
  size_t off = 0;
  auto alloc = [&](size_t bytes) -> char* {
    char* p = ws + off;
    off += (bytes + 255) & ~(size_t)255;
    return p;
  };

  __bf16* inw_bf[4]; __bf16* outw_bf[4];
  for (int i = 0; i < 4; ++i) {
    inw_bf[i]  = (__bf16*)alloc((size_t)1536 * 512 * 2);
    outw_bf[i] = (__bf16*)alloc((size_t)512 * 512 * 2);
  }
  __bf16* w1b = (__bf16*)alloc((size_t)512 * 512 * 2);
  __bf16* w2b = (__bf16*)alloc((size_t)512 * 512 * 2);
  float*  Mf  = (float*)alloc(RD * 4);   __bf16* Mb  = (__bf16*)alloc(RD * 2);
  float*  Ef  = (float*)alloc(RD * 4);   __bf16* Eb  = (__bf16*)alloc(RD * 2);
  float*  QKV = (float*)alloc(R * 1536 * 4);
  __bf16* Qb  = (__bf16*)alloc(RD * 2);
  __bf16* Kb  = (__bf16*)alloc(RD * 2);
  __bf16* Vt  = (__bf16*)alloc(RD * 2);
  float*  scores = (float*)alloc((size_t)128 * 512 * 512 * 4);
  float*  PV  = (float*)alloc(RD * 4);
  __bf16* Acat = (__bf16*)alloc(RD * 2);
  float*  Atmp = (float*)alloc(RD * 4);
  float*  Sbase = (float*)alloc(RD * 4); __bf16* Sbase_b = (__bf16*)alloc(RD * 2);
  __bf16* locA = (__bf16*)alloc((size_t)1024 * 512 * 2);
  float*  locOut = (float*)alloc((size_t)1024 * 512 * 4);
  float*  Sloc = (float*)alloc(RD * 4);  __bf16* Slocb = (__bf16*)alloc(RD * 2);
  float*  Sglob = (float*)alloc(RD * 4);
  float*  Scross = (float*)alloc(RD * 4);
  float*  Ssum = (float*)alloc(RD * 4);  __bf16* Ssum_b = (__bf16*)alloc(RD * 2);
  float*  Hid = (float*)alloc(RD * 4);   __bf16* Hidb = (__bf16*)alloc(RD * 2);
  float*  Ff  = (float*)alloc(RD * 4);
  float*  Fln = (float*)alloc(RD * 4);

  auto conv = [&](const float* src, __bf16* dst, size_t n) {
    k_f32_to_bf16<<<(unsigned)((n + 255) / 256), 256, 0, stream>>>(src, dst, n);
  };
  for (int i = 0; i < 4; ++i) {
    conv(in_w[i],  inw_bf[i],  (size_t)1536 * 512);
    conv(out_w[i], outw_bf[i], (size_t)512 * 512);
  }
  conv(w1, w1b, (size_t)512 * 512);
  conv(w2, w2b, (size_t)512 * 512);

  k_embed<<<(unsigned)((RD + 255) / 256), 256, 0, stream>>>(
      qI, rI, qryI, M_emb, E_emb, Pmat, Mf, Mb, Ef, Eb);

  auto attn = [&](const __bf16* xq, const __bf16* xkv, const __bf16* inw,
                  const float* inb, const __bf16* outw, const float* outbias,
                  int Nq, int causal, float* outC, bool mean_out) {
    const int rows = 16 * Nq;
    const size_t tot = (size_t)rows * 512;
    // Q projection (W rows [0,512)), KV projection (W rows [512,1536))
    k_gemm<<<dim3(rows / 64, 8, 1), 128, 0, stream>>>(
        xq, 512, 0ULL, inw, 512, 0ULL, QKV, 1536, 0ULL, inb, 1.f, 0, 512);
    k_gemm<<<dim3(rows / 64, 16, 1), 128, 0, stream>>>(
        xkv, 512, 0ULL, inw + (size_t)512 * 512, 512, 0ULL, QKV + 512, 1536, 0ULL,
        inb + 512, 1.f, 0, 512);
    k_split_heads<<<(unsigned)((tot + 255) / 256), 256, 0, stream>>>(QKV, Qb, Kb, Vt, Nq, rows);
    // scores = (Q K^T)/8, batched over B*H = 128
    k_gemm<<<dim3(Nq / 64, Nq / 64, 128), 128, 0, stream>>>(
        Qb, 64, (unsigned long long)Nq * 64, Kb, 64, (unsigned long long)Nq * 64,
        scores, Nq, (unsigned long long)Nq * Nq, (const float*)nullptr, 0.125f, 0, 64);
    int sblk = Nq < 256 ? Nq : 256;
    k_softmax<<<128 * Nq, sblk, 0, stream>>>(scores, Nq, Nq, causal);
    if (mean_out)
      k_attn_mean<<<(unsigned)(((size_t)16 * 512 * 512 + 255) / 256), 256, 0, stream>>>(
          (const __bf16*)scores, outF + 8192);
    // O = P V : P bf16 in-place in scores (lda = 2*Nq), W = Vt [64 x Nq]
    k_gemm<<<dim3(Nq / 64, 1, 128), 128, 0, stream>>>(
        (const __bf16*)scores, 2 * Nq, (unsigned long long)Nq * 2 * Nq,
        Vt, Nq, (unsigned long long)64 * Nq,
        PV, 64, (unsigned long long)Nq * 64, (const float*)nullptr, 1.f, 0, Nq);
    k_merge_heads<<<(unsigned)((tot + 255) / 256), 256, 0, stream>>>(PV, Acat, Nq, rows);
    k_gemm<<<dim3(rows / 64, 8, 1), 128, 0, stream>>>(
        Acat, 512, 0ULL, outw, 512, 0ULL, outC, 512, 0ULL, outbias, 1.f, 0, 512);
  };

  // ---- base attention: q=E, k=v=M, causal, attn_w mean output ----
  attn(Eb, Mb, inw_bf[0], in_b[0], outw_bf[0], out_b[0], 512, 1, Atmp, true);
  k_add_ln<<<(unsigned)R, 256, 0, stream>>>(Atmp, Mf, Ef, ln1_g, ln1_b, Sbase, Sbase_b);

  // ---- local attention on last 64 tokens ----
  k_pack_loc<<<(unsigned)(((size_t)1024 * 512 + 255) / 256), 256, 0, stream>>>(Sbase_b, locA);
  attn(locA, locA, inw_bf[1], in_b[1], outw_bf[1], out_b[1], 64, 1, locOut, false);
  k_zero_u32<<<(unsigned)((RD + 255) / 256), 256, 0, stream>>>((uint32_t*)Sloc, RD);
  k_zero_u32<<<(unsigned)((RD / 2 + 255) / 256), 256, 0, stream>>>((uint32_t*)Slocb, RD / 2);
  k_scatter_loc<<<(unsigned)(((size_t)1024 * 512 + 255) / 256), 256, 0, stream>>>(locOut, Sloc, Slocb);

  // ---- global attention (causal) and cross attention (no mask) ----
  attn(Sbase_b, Sbase_b, inw_bf[2], in_b[2], outw_bf[2], out_b[2], 512, 1, Sglob, false);
  attn(Sbase_b, Slocb,   inw_bf[3], in_b[3], outw_bf[3], out_b[3], 512, 0, Scross, false);

  // ---- combine + FFN + LN2 + head ----
  k_add4<<<(unsigned)((RD + 255) / 256), 256, 0, stream>>>(Sbase, Sloc, Sglob, Scross, Ssum, Ssum_b, RD);
  k_gemm<<<dim3((unsigned)(R / 64), 8, 1), 128, 0, stream>>>(
      Ssum_b, 512, 0ULL, w1b, 512, 0ULL, Hid, 512, 0ULL, b1, 1.f, 1, 512);
  conv(Hid, Hidb, RD);
  k_gemm<<<dim3((unsigned)(R / 64), 8, 1), 128, 0, stream>>>(
      Hidb, 512, 0ULL, w2b, 512, 0ULL, Ff, 512, 0ULL, b2, 1.f, 0, 512);
  k_add_ln<<<(unsigned)R, 256, 0, stream>>>(Ff, Ssum, (const float*)nullptr, ln2_g, ln2_b, Fln, (__bf16*)nullptr);
  k_pred<<<(unsigned)(R / 8), 256, 0, stream>>>(Fln, predw, predb, outF);
}
